// SequenceWeight_57406532878973
// MI455X (gfx1250) — compile-verified, matrix operand order checked by reading, not run on hardware
//
#include <hip/hip_runtime.h>

// Problem constants (fixed by the reference)
#define BB   2
#define NN   256
#define LL   384
#define DM   384   // D_MSA == fan_in
#define NH   12    // N_HEAD
#define DH   32    // D_HID
#define FOUT (NH*DH)   // 384

typedef __attribute__((ext_vector_type(16))) __bf16 v16bf;
typedef __attribute__((ext_vector_type(8)))  float  v8f;

// Hardware f32->bf16 (RNE); clang lowers fptrunc to v_cvt_pk_bf16_f32 on gfx1250.
static __device__ __forceinline__ __bf16 f2bf(float x) { return (__bf16)x; }

static __device__ __forceinline__ unsigned short f2bfbits(float x) {
    __bf16 t = (__bf16)x;
    union { __bf16 b; unsigned short s; } u; u.b = t; return u.s;
}

// ---------------------------------------------------------------------------
// Phase 1: per (b,l) compute q = (tar @ Wq^T + bq) * 1/sqrt(32) in LDS, then
// the folded weights w_tilde[h][c] = sum_d q[h*32+d] * Wk[h*32+d, c] -> bf16.
// bk is dropped: its contribution is constant along n and cancels in softmax.
// ---------------------------------------------------------------------------
__global__ void __launch_bounds__(DM)
seqw_phase1(const float* __restrict__ msa, const float* __restrict__ Wq,
            const float* __restrict__ bq,  const float* __restrict__ Wk,
            unsigned short* __restrict__ wtil)
{
    __shared__ float tarS[DM];
    __shared__ float qS[FOUT];

    const int bl = blockIdx.x;          // 0..767
    const int b  = bl / LL;
    const int l  = bl % LL;
    const int tid = threadIdx.x;        // 0..383

    // tar row = msa[b, 0, l, :]
    const float* tar = msa + ((size_t)b * NN * LL + (size_t)l) * DM;
    tarS[tid] = tar[tid];
    __syncthreads();

    // q[tid] over fan_out
    {
        float acc = bq[tid];
        const float* wq = Wq + (size_t)tid * DM;
        #pragma unroll 4
        for (int c = 0; c < DM; ++c) acc += tarS[c] * wq[c];
        qS[tid] = acc * 0.17677669529663687f;   // 1/sqrt(32)
    }
    __syncthreads();

    // w_tilde[h][c], c = tid
    const int c = tid;
    for (int h = 0; h < NH; ++h) {
        const float* qh = qS + h * DH;
        float acc = 0.0f;
        #pragma unroll
        for (int d = 0; d < DH; ++d)
            acc += qh[d] * Wk[(size_t)(h * DH + d) * DM + c];
        wtil[((size_t)bl * NH + h) * DM + c] = f2bfbits(acc);
    }
}

// ---------------------------------------------------------------------------
// Phase 2: per (b,l) block (256 thr = 8 waves):
//   attn[n,h] = msa[b,n,l,:] . w_tilde[b,l,h,:]   via v_wmma_f32_16x16x32_bf16
//   then fused softmax over n (wave32 shfl reductions), write (b,n,l,h).
// Each msa element is read exactly once -> HBM-bound (~13 us at 23.3 TB/s).
// ---------------------------------------------------------------------------
__global__ void __launch_bounds__(256)
seqw_phase2(const float* __restrict__ msa, const unsigned short* __restrict__ wtil,
            float* __restrict__ out)
{
    __shared__ unsigned short wS[16 * DM];   // 16 heads (12 valid, 4 zero) x 384, bf16 bits
    __shared__ float attnS[NN * 16];         // 256 n x 16 h (f32)

    const int bl  = blockIdx.x;
    const int b   = bl / LL;
    const int l   = bl % LL;
    const int tid = threadIdx.x;
    const int lane = tid & 31;
    const int wave = tid >> 5;

    // Stage padded w_tilde into LDS
    for (int idx = tid; idx < 16 * DM; idx += 256) {
        const int h = idx / DM, c = idx % DM;
        wS[idx] = (h < NH) ? wtil[((size_t)bl * NH + h) * DM + c] : (unsigned short)0;
    }
    __syncthreads();

    const int hcol  = lane & 15;   // N column of the WMMA tile (head)
    const int khalf = lane >> 4;   // which K half this lane carries

    // Preload all 12 B fragments (K-steps of 32 over c=384), shared by both n-tiles.
    // 16-bit B layout mirrors A: lane holds col hcol, K = {0..7,16..23}+8*khalf.
    union BU { v16bf v; unsigned short s[16]; };
    BU Bf[12];
    #pragma unroll
    for (int k = 0; k < 12; ++k) {
        const int base0 = hcol * DM + k * 32 + 8 * khalf;
        const int base1 = base0 + 16;
        #pragma unroll
        for (int e = 0; e < 8; ++e) {
            Bf[k].s[e]     = wS[base0 + e];
            Bf[k].s[8 + e] = wS[base1 + e];
        }
    }

    const int m = lane & 15;       // A row within tile
    #pragma unroll
    for (int tt = 0; tt < 2; ++tt) {
        const int t = wave + tt * 8;        // n-tile 0..15
        const int n = t * 16 + m;
        const float* rowp = msa + (((size_t)b * NN + n) * LL + l) * DM;

        v8f acc = {};
        #pragma unroll
        for (int k = 0; k < 12; ++k) {
            // Two contiguous 32B runs per lane, matching the 16-bit A VGPR layout.
            const float4* p0 = (const float4*)(rowp + k * 32 + 8 * khalf);
            const float4* p1 = (const float4*)(rowp + k * 32 + 16 + 8 * khalf);
            const float4 x0 = p0[0], x1 = p0[1];
            const float4 y0 = p1[0], y1 = p1[1];
            union AU { v16bf v; __bf16 e[16]; } a;
            a.e[0]  = f2bf(x0.x); a.e[1]  = f2bf(x0.y);
            a.e[2]  = f2bf(x0.z); a.e[3]  = f2bf(x0.w);
            a.e[4]  = f2bf(x1.x); a.e[5]  = f2bf(x1.y);
            a.e[6]  = f2bf(x1.z); a.e[7]  = f2bf(x1.w);
            a.e[8]  = f2bf(y0.x); a.e[9]  = f2bf(y0.y);
            a.e[10] = f2bf(y0.z); a.e[11] = f2bf(y0.w);
            a.e[12] = f2bf(y1.x); a.e[13] = f2bf(y1.y);
            a.e[14] = f2bf(y1.z); a.e[15] = f2bf(y1.w);
            acc = __builtin_amdgcn_wmma_f32_16x16x32_bf16(
                false, a.v, false, Bf[k].v, (short)0, acc, false, false);
        }
        // C/D layout: lane -> col hcol; element r -> row r + 8*khalf
        #pragma unroll
        for (int r = 0; r < 8; ++r)
            attnS[(t * 16 + r + 8 * khalf) * 16 + hcol] = acc[r];
    }
    __syncthreads();

    // Fused softmax over n (256) per head; wave w handles heads {w, w+8} (<12).
    for (int h = wave; h < NH; h += 8) {
        float vals[8];
        float mx = -3.402823466e38f;
        #pragma unroll
        for (int i = 0; i < 8; ++i) {
            vals[i] = attnS[(lane * 8 + i) * 16 + h];
            mx = fmaxf(mx, vals[i]);
        }
        for (int off = 16; off > 0; off >>= 1)
            mx = fmaxf(mx, __shfl_xor(mx, off, 32));
        float s = 0.0f;
        #pragma unroll
        for (int i = 0; i < 8; ++i) { vals[i] = __expf(vals[i] - mx); s += vals[i]; }
        for (int off = 16; off > 0; off >>= 1)
            s += __shfl_xor(s, off, 32);
        const float inv = 1.0f / s;
        #pragma unroll
        for (int i = 0; i < 8; ++i) {
            const int n = lane * 8 + i;
            out[(((size_t)b * NN + n) * LL + l) * NH + h] = vals[i] * inv;
        }
    }
}

extern "C" void kernel_launch(void* const* d_in, const int* in_sizes, int n_in,
                              void* d_out, int out_size, void* d_ws, size_t ws_size,
                              hipStream_t stream) {
    (void)in_sizes; (void)n_in; (void)out_size; (void)ws_size;
    const float* msa = (const float*)d_in[0];
    const float* Wq  = (const float*)d_in[1];
    const float* bq  = (const float*)d_in[2];
    const float* Wk  = (const float*)d_in[3];
    // d_in[4] = bk: constant along the softmax axis -> cancels, intentionally unused.

    float* out = (float*)d_out;
    unsigned short* wtil = (unsigned short*)d_ws;  // BB*LL*NH*DM bf16 = ~7.1 MB

    seqw_phase1<<<BB * LL, DM, 0, stream>>>(msa, Wq, bq, Wk, wtil);
    seqw_phase2<<<BB * LL, 256, 0, stream>>>(msa, wtil, out);
}
